// AttentionDecoder_51668456571148
// MI455X (gfx1250) — matrix operand order repaired and checked by reading, not compiled, not run on hardware
//
#include <hip/hip_runtime.h>
#include <math.h>

// Problem constants (from reference): B=64, L=480, C=H=E=512, V=8000, T-1=128
#define B_ 64
#define L_ 480
#define C_ 512
#define H_ 512
#define E_ 512
#define V_ 8000
#define TSTEPS 128
#define KCAT 1536   // E + C + H
#define NGATES 2048 // 4*H

typedef __attribute__((ext_vector_type(16))) __bf16 v16bf;
typedef __attribute__((ext_vector_type(8)))  __bf16 v8bf;
typedef __attribute__((ext_vector_type(8)))  float  v8f;

__device__ __forceinline__ unsigned short f2bf(float f) {
  unsigned u = __float_as_uint(f);
  u += 0x7fffu + ((u >> 16) & 1u);           // round-to-nearest-even
  return (unsigned short)(u >> 16);
}
__device__ __forceinline__ float bf2f(unsigned short h) {
  return __uint_as_float(((unsigned)h) << 16);
}
__device__ __forceinline__ float sigm_(float x) { return 1.f / (1.f + __expf(-x)); }

// ---------------------------------------------------------------------------
// Generic skinny GEMM on matrix cores:  Cf(64 x N) = A(64 x K,bf16) * W^T
// W is (N x K) row-major bf16 (PyTorch/JAX weight layout => natural WMMA B).
// Each wave: one 16-row M tile x four 16-col N tiles, K in steps of 32.
// Optional bias, optional tanh, optional bf16 mirror stores (two targets).
// ---------------------------------------------------------------------------
__global__ __launch_bounds__(256) void k_gemm_wmma(
    const unsigned short* __restrict__ A, int lda,
    const unsigned short* __restrict__ W,
    const float* __restrict__ bias,
    float* __restrict__ Cf,
    unsigned short* __restrict__ bf0, int ldbf0,
    unsigned short* __restrict__ bf1, int ldbf1,
    int N, int K, int do_tanh)
{
  const int lane = threadIdx.x & 31;
  const int gw   = (blockIdx.x * blockDim.x + threadIdx.x) >> 5;
  const int nquads = N >> 6;              // N-tiles of 64 columns
  const int mtile  = gw & 3;              // 64 rows = 4 M-tiles
  const int quad   = gw >> 2;
  if (quad >= nquads) return;

  const int m  = lane & 15;
  const int hi = lane >> 4;
  // A fragment (16x32 bf16): lane m holds k = 16*hi .. +16
  const unsigned short* arow = A + (size_t)(mtile * 16 + m) * lda + 16 * hi;
  // B fragment (32x16 bf16): lane holds col n=(lane&15), k = 16*hi .. +16
  const unsigned short* wbase = W + (size_t)m * K + 16 * hi; // + n-tile*16*K added per j

  v8f acc[4];
  const v8f vz = {0.f, 0.f, 0.f, 0.f, 0.f, 0.f, 0.f, 0.f};
#pragma unroll
  for (int j = 0; j < 4; ++j) acc[j] = vz;

  for (int k = 0; k < K; k += 32) {
    union { v16bf v; v8bf h[2]; } ua;
    ua.h[0] = *(const v8bf*)(arow + k);
    ua.h[1] = *(const v8bf*)(arow + k + 8);
    const v16bf afrag = ua.v;
    __builtin_prefetch(arow + k + 256, 0, 1);   // -> global_prefetch_b8
#pragma unroll
    for (int j = 0; j < 4; ++j) {
      const unsigned short* wp = wbase + (size_t)((quad * 4 + j) * 16) * K + k;
      const v16bf bfrag = *(const v16bf*)wp;
      acc[j] = __builtin_amdgcn_wmma_f32_16x16x32_bf16(
          false, afrag, false, bfrag, (short)0, acc[j], false, false);
    }
  }

#pragma unroll
  for (int j = 0; j < 4; ++j) {
    const int nc = (quad * 4 + j) * 16 + m;
    const float bv = bias ? bias[nc] : 0.f;
#pragma unroll
    for (int r = 0; r < 8; ++r) {
      const int mr = mtile * 16 + 8 * hi + r;   // C layout: VGPR r, lanes>=16 -> M+8
      float v = acc[j][r] + bv;
      if (do_tanh) v = tanhf(v);
      Cf[(size_t)mr * N + nc] = v;
      if (bf0) bf0[(size_t)mr * ldbf0 + nc] = f2bf(v);
      if (bf1) bf1[(size_t)mr * ldbf1 + nc] = f2bf(v);
    }
  }
}

// ----------------------------- prep kernels --------------------------------
__global__ void k_f2bf(const float* __restrict__ s, unsigned short* __restrict__ d, long n) {
  long i = (long)blockIdx.x * blockDim.x + threadIdx.x;
  const long st = (long)gridDim.x * blockDim.x;
  for (; i < n; i += st) d[i] = f2bf(s[i]);
}

__global__ void k_wcat(const float* __restrict__ Wih, const float* __restrict__ Whh,
                       unsigned short* __restrict__ wcat) {
  long i = (long)blockIdx.x * blockDim.x + threadIdx.x;
  const long st = (long)gridDim.x * blockDim.x;
  const long total = (long)NGATES * KCAT;
  for (; i < total; i += st) {
    const int n = (int)(i / KCAT), k = (int)(i % KCAT);
    const float v = (k < E_ + C_) ? Wih[(size_t)n * (E_ + C_) + k]
                                  : Whh[(size_t)n * H_ + (k - (E_ + C_))];
    wcat[i] = f2bf(v);
  }
}

__global__ void k_embed(const float* __restrict__ embW, const int* __restrict__ seq,
                        unsigned short* __restrict__ xs) {
  long i = (long)blockIdx.x * blockDim.x + threadIdx.x;
  const long st = (long)gridDim.x * blockDim.x;
  const long total = (long)B_ * TSTEPS * E_;
  for (; i < total; i += st) {
    const int e = (int)(i & (E_ - 1));
    const int t = (int)((i >> 9) & (TSTEPS - 1));
    const int b = (int)(i >> 16);
    const int tok = seq[b * 129 + t];      // seq[:, :-1]
    xs[i] = f2bf(embW[(size_t)tok * E_ + e]);
  }
}

// ----------------------------- per-step kernels ----------------------------
__global__ __launch_bounds__(512) void k_packx(const unsigned short* __restrict__ xs,
                                               unsigned short* __restrict__ inp, int t) {
  const int b = blockIdx.x, e = threadIdx.x;
  inp[(size_t)b * KCAT + e] = xs[((size_t)b * TSTEPS + t) * E_ + e];
}

__global__ __launch_bounds__(512) void k_lstm(const float* __restrict__ gates,
                                              const float* __restrict__ bih,
                                              const float* __restrict__ bhh,
                                              float* __restrict__ cst,
                                              unsigned short* __restrict__ h2bf,
                                              unsigned short* __restrict__ inp,
                                              unsigned short* __restrict__ inp2) {
  const int b = blockIdx.x, n = threadIdx.x;
  const float* g = gates + (size_t)b * NGATES;
  const float i = sigm_(g[n]            + bih[n]            + bhh[n]);
  const float f = sigm_(g[H_ + n]       + bih[H_ + n]       + bhh[H_ + n]);
  const float gg = tanhf(g[2 * H_ + n]  + bih[2 * H_ + n]   + bhh[2 * H_ + n]);
  const float o = sigm_(g[3 * H_ + n]   + bih[3 * H_ + n]   + bhh[3 * H_ + n]);
  const float c2 = f * cst[(size_t)b * H_ + n] + i * gg;
  const float h2 = o * tanhf(c2);
  cst[(size_t)b * H_ + n] = c2;
  const unsigned short hb = f2bf(h2);
  h2bf[(size_t)b * H_ + n] = hb;
  inp[(size_t)b * KCAT + (E_ + C_) + n] = hb;   // h slot for next step's gates
  inp2[(size_t)b * (C_ + H_) + C_ + n] = hb;    // h2 slot for W_om input
}

__global__ __launch_bounds__(256) void k_attn(const unsigned short* __restrict__ cnnbf,
                                              const float* __restrict__ mapped,
                                              float* __restrict__ attn) {
  const int b = blockIdx.x;
  const int wave = threadIdx.x >> 5, lane = threadIdx.x & 31;
  const int l = blockIdx.y * 8 + wave;          // 60 * 8 = 480
  const unsigned int* row = (const unsigned int*)(cnnbf + ((size_t)b * L_ + l) * C_);
  const float2* mp = (const float2*)(mapped + (size_t)b * C_);
  float acc = 0.f;
#pragma unroll
  for (int i = 0; i < 8; ++i) {                 // 16 channels per lane
    const unsigned int pk = row[lane * 8 + i];
    const float2 mv = mp[lane * 8 + i];
    acc += bf2f((unsigned short)pk) * mv.x + bf2f((unsigned short)(pk >> 16)) * mv.y;
  }
#pragma unroll
  for (int off = 16; off >= 1; off >>= 1) acc += __shfl_xor(acc, off, 32);
  if (lane == 0) attn[(size_t)b * L_ + l] = acc;
}

__global__ __launch_bounds__(512) void k_softmax(const float* __restrict__ attn,
                                                 float* __restrict__ w) {
  __shared__ float red[512];
  const int b = blockIdx.x, t = threadIdx.x;
  const float x = (t < L_) ? attn[(size_t)b * L_ + t] : -3.4e38f;
  red[t] = x; __syncthreads();
  for (int s = 256; s > 0; s >>= 1) { if (t < s) red[t] = fmaxf(red[t], red[t + s]); __syncthreads(); }
  const float mx = red[0]; __syncthreads();
  const float e = (t < L_) ? __expf(x - mx) : 0.f;
  red[t] = e; __syncthreads();
  for (int s = 256; s > 0; s >>= 1) { if (t < s) red[t] += red[t + s]; __syncthreads(); }
  const float inv = 1.f / red[0];
  if (t < L_) w[(size_t)b * L_ + t] = e * inv;
}

__global__ __launch_bounds__(512) void k_ctx(const unsigned short* __restrict__ cnnbf,
                                             const float* __restrict__ w,
                                             unsigned short* __restrict__ inp2) {
  const int b = blockIdx.x, c = threadIdx.x;
  float acc = 0.f;
  const unsigned short* base = cnnbf + (size_t)b * L_ * C_ + c;
  const float* wb = w + (size_t)b * L_;
  for (int l = 0; l < L_; ++l) acc += wb[l] * bf2f(base[(size_t)l * C_]);
  inp2[(size_t)b * (C_ + H_) + c] = f2bf(acc);  // ctx_comb slot for W_om input
}

__global__ __launch_bounds__(1024) void k_logsoftmax(const float* __restrict__ logits,
                                                     float* __restrict__ out, int t) {
  __shared__ float red[1024];
  const int b = blockIdx.x, tid = threadIdx.x;
  const float* row = logits + (size_t)b * V_;
  float mx = -3.4e38f;
  for (int v = tid; v < V_; v += 1024) mx = fmaxf(mx, row[v]);
  red[tid] = mx; __syncthreads();
  for (int s = 512; s > 0; s >>= 1) { if (tid < s) red[tid] = fmaxf(red[tid], red[tid + s]); __syncthreads(); }
  mx = red[0]; __syncthreads();
  float sum = 0.f;
  for (int v = tid; v < V_; v += 1024) sum += __expf(row[v] - mx);
  red[tid] = sum; __syncthreads();
  for (int s = 512; s > 0; s >>= 1) { if (tid < s) red[tid] += red[tid + s]; __syncthreads(); }
  const float lse = mx + __logf(red[0]);
  float* orow = out + ((size_t)b * TSTEPS + t) * V_;
  for (int v = tid; v < V_; v += 1024) orow[v] = row[v] - lse;
}

// ---------------------------------------------------------------------------
extern "C" void kernel_launch(void* const* d_in, const int* in_sizes, int n_in,
                              void* d_out, int out_size, void* d_ws, size_t ws_size,
                              hipStream_t stream) {
  (void)in_sizes; (void)n_in; (void)out_size; (void)ws_size;
  const float* cnn     = (const float*)d_in[0];
  const int*   seq     = (const int*)d_in[1];
  const float* embW    = (const float*)d_in[2];
  const float* W_ih    = (const float*)d_in[3];
  const float* b_ih    = (const float*)d_in[4];
  const float* W_hh    = (const float*)d_in[5];
  const float* b_hh    = (const float*)d_in[6];
  const float* W_hm    = (const float*)d_in[7];
  const float* W_om    = (const float*)d_in[8];
  const float* W_logit = (const float*)d_in[9];
  const float* b_logit = (const float*)d_in[10];
  float* out = (float*)d_out;

  // Workspace carve-out (~60 MB total; 192MB L2 keeps all of it hot)
  char* ws = (char*)d_ws;
  size_t off = 0;
  auto carve = [&](size_t bytes) -> char* {
    char* p = ws + off;
    off = (off + bytes + 255) & ~(size_t)255;
    return p;
  };
  unsigned short* Wcat_bf   = (unsigned short*)carve((size_t)NGATES * KCAT * 2);
  unsigned short* Whm_bf    = (unsigned short*)carve((size_t)C_ * H_ * 2);
  unsigned short* Wom_bf    = (unsigned short*)carve((size_t)H_ * (C_ + H_) * 2);
  unsigned short* Wlogit_bf = (unsigned short*)carve((size_t)V_ * H_ * 2);
  unsigned short* cnn_bf    = (unsigned short*)carve((size_t)B_ * L_ * C_ * 2);
  unsigned short* xs_bf     = (unsigned short*)carve((size_t)B_ * TSTEPS * E_ * 2);
  unsigned short* inp_bf    = (unsigned short*)carve((size_t)B_ * KCAT * 2);
  unsigned short* inp2_bf   = (unsigned short*)carve((size_t)B_ * (C_ + H_) * 2);
  unsigned short* h2_bf     = (unsigned short*)carve((size_t)B_ * H_ * 2);
  unsigned short* out_bf    = (unsigned short*)carve((size_t)B_ * H_ * 2);
  float* c_f32      = (float*)carve((size_t)B_ * H_ * 4);
  float* gates_f32  = (float*)carve((size_t)B_ * NGATES * 4);
  float* mapped_f32 = (float*)carve((size_t)B_ * H_ * 4);
  float* attn_f32   = (float*)carve((size_t)B_ * L_ * 4);
  float* w_f32      = (float*)carve((size_t)B_ * L_ * 4);
  float* outC_f32   = (float*)carve((size_t)B_ * H_ * 4);
  float* logits_f32 = (float*)carve((size_t)B_ * V_ * 4);

  // ---- one-time prep (re-run each call: deterministic, no static state) ----
  auto gsblocks = [](long n) { long b = (n + 255) / 256; return (int)(b > 16384 ? 16384 : b); };
  k_wcat<<<gsblocks((long)NGATES * KCAT), 256, 0, stream>>>(W_ih, W_hh, Wcat_bf);
  k_f2bf<<<gsblocks((long)C_ * H_), 256, 0, stream>>>(W_hm, Whm_bf, (long)C_ * H_);
  k_f2bf<<<gsblocks((long)H_ * (C_ + H_)), 256, 0, stream>>>(W_om, Wom_bf, (long)H_ * (C_ + H_));
  k_f2bf<<<gsblocks((long)V_ * H_), 256, 0, stream>>>(W_logit, Wlogit_bf, (long)V_ * H_);
  k_f2bf<<<gsblocks((long)B_ * L_ * C_), 256, 0, stream>>>(cnn, cnn_bf, (long)B_ * L_ * C_);
  k_embed<<<gsblocks((long)B_ * TSTEPS * E_), 256, 0, stream>>>(embW, seq, xs_bf);
  hipMemsetAsync(inp_bf, 0, (size_t)B_ * KCAT * 2, stream);  // zero ctx/h carry
  hipMemsetAsync(c_f32, 0, (size_t)B_ * H_ * 4, stream);     // zero cell state

  auto gemm_blocks = [](int N) { return (4 * (N >> 6) + 7) / 8; };

  // ---- 128 sequential decoder steps ----
  for (int t = 0; t < TSTEPS; ++t) {
    k_packx<<<B_, 512, 0, stream>>>(xs_bf, inp_bf, t);

    // gates = [xt|ctx|h] @ Wcat^T   (M=64, N=2048, K=1536)
    k_gemm_wmma<<<gemm_blocks(NGATES), 256, 0, stream>>>(
        inp_bf, KCAT, Wcat_bf, nullptr, gates_f32,
        nullptr, 0, nullptr, 0, NGATES, KCAT, 0);

    k_lstm<<<B_, 512, 0, stream>>>(gates_f32, b_ih, b_hh, c_f32, h2_bf, inp_bf, inp2_bf);

    // mapped = h2 @ W_hm^T   (M=64, N=512, K=512)
    k_gemm_wmma<<<gemm_blocks(C_), 256, 0, stream>>>(
        h2_bf, H_, Whm_bf, nullptr, mapped_f32,
        nullptr, 0, nullptr, 0, C_, H_, 0);

    k_attn<<<dim3(B_, L_ / 8), 256, 0, stream>>>(cnn_bf, mapped_f32, attn_f32);
    k_softmax<<<B_, 512, 0, stream>>>(attn_f32, w_f32);
    k_ctx<<<B_, 512, 0, stream>>>(cnn_bf, w_f32, inp2_bf);

    // out = tanh([ctx_comb|h2] @ W_om^T); mirrors: out_bf (logit A) and
    // inp_bf[:,512:1024] (the "ctx" carried into the next step's gates)
    k_gemm_wmma<<<gemm_blocks(H_), 256, 0, stream>>>(
        inp2_bf, C_ + H_, Wom_bf, nullptr, outC_f32,
        out_bf, H_, inp_bf + E_, KCAT, H_, C_ + H_, 1);

    // logits = out @ W_logit^T + b_logit   (M=64, N=8000, K=512)
    k_gemm_wmma<<<gemm_blocks(V_), 256, 0, stream>>>(
        out_bf, H_, Wlogit_bf, b_logit, logits_f32,
        nullptr, 0, nullptr, 0, V_, H_, 0);

    k_logsoftmax<<<B_, 1024, 0, stream>>>(logits_f32, out, t);
  }
}
